// ContrastiveLoss_69252052680737
// MI455X (gfx1250) — matrix operand order repaired and checked by reading, not compiled, not run on hardware
//
#include <hip/hip_runtime.h>
#include <math.h>

// ---------------------------------------------------------------------------
// ContrastiveLoss on MI455X (gfx1250, wave32).
//
// Gram matrices are the hot loop (~6.4 GFLOP, operands fully L2-resident at
// 2 MB), so we run them on the bf16 matrix pipe with the split-bf16 trick:
//   x = hi(x) + lo(x)  (both bf16)
//   x . y ~= hi.hi + hi.lo + lo.hi      (error ~2^-17, fp32-comparable)
// Each wave computes a 16x16 Gram tile with 6 chained
// v_wmma_f32_16x16x32_bf16 (2 K-blocks x 3 split terms) instead of 16 slow
// f32 16x16x4 WMMAs. Elementwise gate/bell math runs on VALU, co-executing
// with the XDL pipe; histogram uses LDS ds-atomics; reductions use
// global_atomic_add_f32.
// ---------------------------------------------------------------------------

typedef __attribute__((ext_vector_type(16))) __bf16 v16bf;
typedef __attribute__((ext_vector_type(8)))  __bf16 v8bf;
typedef __attribute__((ext_vector_type(8)))  float  v8f;

#define N_PTS   4096
#define DIMK    64
#define NELEM   (N_PTS * DIMK)
#define NPAIRS  8386560.0f   // 4096*4095/2
#define NBINS   1024
#define BINW    0.03125f     // 32.0 / 1024 distance range

// workspace layout:
//  wsu[0..1023]    : distance histogram (uint)
//  wsf[1024]       : sum exp(-(d-2)^2/18)
//  wsf[1025]       : sum relu(lower - d)
//  wsf[1026]       : sum exp(-dsq/2) codebook
//  wsf[1027..1030] : lower, upper, center, sigma
//  wsf[1031+p]     : S_w[p]   p<10
//  wsf[1041+p]     : S_wb[p]  p<10
//  wsf[2048..6143] : row norms of z
//  wsf[6144..10239]: row norms of codebook
//  byte 40960+     : Hz, Lz, Hc, Lc  (bf16 hi/lo planes, 512 KB each)

static __device__ const int g_bi[10] = {0,0,0,0,1,1,1,2,2,3};
static __device__ const int g_bj[10] = {0,1,2,3,1,2,3,2,3,3};

__device__ __forceinline__ float fast_sigmoid(float x) {
  return 1.0f / (1.0f + __expf(-x));
}

#define SHUF16(a, b) __builtin_shufflevector(a, b, 0,1,2,3,4,5,6,7,8,9,10,11,12,13,14,15)

// 16x16 Gram tile G[m][n] = sum_k X[row0+m][k] * X[col0+n][k]  (split bf16)
__device__ __forceinline__ v8f gram_tile(const __bf16* __restrict__ H,
                                         const __bf16* __restrict__ L,
                                         int row0, int col0, int lane) {
  const int half = lane >> 4;
  const int l    = lane & 15;
  const int arow = (row0 + l) * DIMK;
  const int brow = (col0 + l) * DIMK;
  const int aoff = half * 16;   // A k order used here: contiguous 16 at kb+16h (matches B)
  const int boff = half * 16;   // B k order used here: contiguous 16 at kb+16h (matches A)
  v8f acc = {};
#pragma unroll
  for (int kb = 0; kb < DIMK; kb += 32) {
    const v8bf ah0 = *(const v8bf*)(H + arow + kb + aoff);
    const v8bf ah1 = *(const v8bf*)(H + arow + kb + aoff + 8);
    const v8bf al0 = *(const v8bf*)(L + arow + kb + aoff);
    const v8bf al1 = *(const v8bf*)(L + arow + kb + aoff + 8);
    const v8bf bh0 = *(const v8bf*)(H + brow + kb + boff);
    const v8bf bh1 = *(const v8bf*)(H + brow + kb + boff + 8);
    const v8bf bl0 = *(const v8bf*)(L + brow + kb + boff);
    const v8bf bl1 = *(const v8bf*)(L + brow + kb + boff + 8);
    const v16bf Ah = SHUF16(ah0, ah1);
    const v16bf Al = SHUF16(al0, al1);
    const v16bf Bh = SHUF16(bh0, bh1);
    const v16bf Bl = SHUF16(bl0, bl1);
    acc = __builtin_amdgcn_wmma_f32_16x16x32_bf16(false, Ah, false, Bh,
                                                  (short)0, acc, false, false);
    acc = __builtin_amdgcn_wmma_f32_16x16x32_bf16(false, Ah, false, Bl,
                                                  (short)0, acc, false, false);
    acc = __builtin_amdgcn_wmma_f32_16x16x32_bf16(false, Al, false, Bh,
                                                  (short)0, acc, false, false);
  }
  return acc;
}

// ---------------------------------------------------------------------------
__global__ void zero_ws_kernel(unsigned* wsu) {
  int i = blockIdx.x * blockDim.x + threadIdx.x;
  for (; i < 1056; i += gridDim.x * blockDim.x) wsu[i] = 0u;
}

// fp32 -> (hi, lo) bf16 split planes
__global__ void split_bf16_kernel(const float* __restrict__ X,
                                  __bf16* __restrict__ H,
                                  __bf16* __restrict__ L) {
  const int i = blockIdx.x * blockDim.x + threadIdx.x;
  if (i < NELEM) {
    const float x = X[i];
    const __bf16 h = (__bf16)x;
    H[i] = h;
    L[i] = (__bf16)(x - (float)h);
  }
}

// one wave per row: |x_row|^2 (exact fp32)
__global__ void row_norms_kernel(const float* __restrict__ X, float* __restrict__ norms) {
  const int wave = threadIdx.x >> 5;
  const int lane = threadIdx.x & 31;
  const int row  = blockIdx.x * 8 + wave;
  if (row >= N_PTS) return;
  const float a = X[row * DIMK + lane];
  const float b = X[row * DIMK + 32 + lane];
  float s = a * a + b * b;
#pragma unroll
  for (int m = 16; m >= 1; m >>= 1) s += __shfl_xor(s, m, 32);
  if (lane == 0) norms[row] = s;
}

// ---------------------------------------------------------------------------
// mode 0: z, global triu  -> LDS histogram + sum exp(-(d-2)^2/18)
// mode 2: z, global triu  -> sum relu(lower - d)   (lower from wsf[1027])
// mode 3: codebook, triu  -> sum exp(-dsq/2)
__global__ void pair_tile_kernel(const __bf16* __restrict__ H,
                                 const __bf16* __restrict__ L,
                                 const float* __restrict__ norms,
                                 float* __restrict__ wsf,
                                 unsigned* __restrict__ wsu,
                                 int mode) {
  __shared__ unsigned hist[NBINS];
  const int tid = threadIdx.x;
  if (mode == 0) {
    for (int b = tid; b < NBINS; b += 256) hist[b] = 0u;
  }
  __syncthreads();

  const int wave = tid >> 5;
  const int lane = tid & 31;
  const int t    = blockIdx.x * 8 + wave;     // 256x256 tile grid
  const int ti   = t >> 8;
  const int tj   = t & 255;

  float partial = 0.0f;
  if (ti <= tj) {
    const int row0 = ti * 16, col0 = tj * 16;
    const int half = lane >> 4, l = lane & 15;
    const v8f acc  = gram_tile(H, L, row0, col0, lane);
    const int gj   = col0 + l;
    const float nj = norms[gj];
    const float lower = wsf[1027];
#pragma unroll
    for (int r = 0; r < 8; ++r) {
      const int gi = row0 + half * 8 + r;     // C-layout: lanes>=16 hold M=8..15
      if (gi < gj) {
        const float dsq = fmaxf(norms[gi] + nj - 2.0f * acc[r], 0.0f);
        if (mode == 3) {
          partial += __expf(-0.5f * dsq);
        } else {
          const float d = sqrtf(fmaxf(dsq, 1e-12f));
          if (mode == 0) {
            int b = (int)(d * (1.0f / BINW));
            b = b > (NBINS - 1) ? (NBINS - 1) : b;
            atomicAdd(&hist[b], 1u);
            partial += __expf(-(d - 2.0f) * (d - 2.0f) * (1.0f / 18.0f));
          } else {  // mode 2
            partial += fmaxf(lower - d, 0.0f);
          }
        }
      }
    }
  }
#pragma unroll
  for (int m = 16; m >= 1; m >>= 1) partial += __shfl_xor(partial, m, 32);
  if (lane == 0 && partial != 0.0f) {
    const int idx = (mode == 0) ? 1024 : (mode == 2) ? 1025 : 1026;
    atomicAdd(&wsf[idx], partial);
  }
  if (mode == 0) {
    __syncthreads();
    for (int b = tid; b < NBINS; b += 256) {
      const unsigned h = hist[b];
      if (h) atomicAdd(&wsu[b], h);
    }
  }
}

// ---------------------------------------------------------------------------
__global__ void quantile_kernel(float* wsf, const unsigned* wsu) {
  if (threadIdx.x != 0) return;
  unsigned total = 0;
  for (int b = 0; b < NBINS; ++b) total += wsu[b];
  const float t25 = 0.25f * (float)total;
  const float t75 = 0.75f * (float)total;
  float cum = 0.0f, lower = 0.0f, upper = 0.0f;
  for (int b = 0; b < NBINS; ++b) {
    const float c = (float)wsu[b];
    if (cum < t25 && cum + c >= t25)
      lower = ((float)b + (t25 - cum) / fmaxf(c, 1.0f)) * BINW;
    if (cum < t75 && cum + c >= t75)
      upper = ((float)b + (t75 - cum) / fmaxf(c, 1.0f)) * BINW;
    cum += c;
  }
  const float band = fabsf(upper - lower);
  float sigma = fmaxf(1e-4f, 0.2f * band);
  sigma = fmaxf(sigma, 1e-6f);
  wsf[1027] = lower;
  wsf[1028] = upper;
  wsf[1029] = 0.5f * (lower + upper);
  wsf[1030] = sigma;
}

// ---------------------------------------------------------------------------
// Mid loss: 10 block pairs (bi<=bj), block-local triu mask applied to every
// block pair (faithful to the reference quirk). grid.y = pair index.
__global__ void mid_tile_kernel(const __bf16* __restrict__ H,
                                const __bf16* __restrict__ L,
                                const float* __restrict__ norms,
                                float* __restrict__ wsf) {
  const int wave = threadIdx.x >> 5;
  const int lane = threadIdx.x & 31;
  const int t    = blockIdx.x * 8 + wave;     // 64x64 local tile grid
  const int lti  = t >> 6;
  const int ltj  = t & 63;
  const int p    = blockIdx.y;

  float sw = 0.0f, swb = 0.0f;
  if (lti <= ltj) {
    const int bi = g_bi[p], bj = g_bj[p];
    const int row0 = bi * 1024 + lti * 16;
    const int col0 = bj * 1024 + ltj * 16;
    const float lower  = wsf[1027];
    const float upper  = wsf[1028];
    const float center = wsf[1029];
    const float sigma  = wsf[1030];
    const float inv2s2 = 1.0f / (2.0f * sigma * sigma);

    const int half = lane >> 4, l = lane & 15;
    const v8f acc  = gram_tile(H, L, row0, col0, lane);
    const int gj   = col0 + l;
    const int lj   = ltj * 16 + l;            // block-local column
    const float nj = norms[gj];
#pragma unroll
    for (int r = 0; r < 8; ++r) {
      const int li = lti * 16 + half * 8 + r; // block-local row
      if (li < lj) {
        const int gi = row0 + half * 8 + r;
        const float dsq = fmaxf(norms[gi] + nj - 2.0f * acc[r], 0.0f);
        const float d   = sqrtf(fmaxf(dsq, 1e-12f));
        const float x1  = fminf(fmaxf(20.0f * (d - lower), -40.0f), 40.0f);
        const float x2  = fminf(fmaxf(20.0f * (upper - d), -40.0f), 40.0f);
        const float w   = fast_sigmoid(x1) * fast_sigmoid(x2);
        const float e   = fmaxf(-(d - center) * (d - center) * inv2s2, -60.0f);
        sw  += w;
        swb += w * __expf(e);
      }
    }
  }
#pragma unroll
  for (int m = 16; m >= 1; m >>= 1) {
    sw  += __shfl_xor(sw, m, 32);
    swb += __shfl_xor(swb, m, 32);
  }
  if (lane == 0) {
    if (sw  != 0.0f) atomicAdd(&wsf[1031 + p], sw);
    if (swb != 0.0f) atomicAdd(&wsf[1041 + p], swb);
  }
}

// ---------------------------------------------------------------------------
__global__ void finalize_kernel(const float* wsf, float* out) {
  if (threadIdx.x != 0 || blockIdx.x != 0) return;
  const float repel2    = wsf[1024] / NPAIRS;
  const float relu_mean = wsf[1025] / NPAIRS;
  const float cb_loss   = wsf[1026] / NPAIRS;
  float mid = 0.0f;
  for (int p = 0; p < 10; ++p) {
    const float den = fmaxf(wsf[1031 + p], 1e-8f);
    float v = wsf[1041 + p] / den;
    if (!(v == v) || fabsf(v) > 3.0e38f) v = 0.0f;   // nan_to_num
    mid += v;
  }
  mid *= 0.1f;
  out[0] = mid;                 // final_loss
  out[1] = 1.0f;                // neg_loss
  out[2] = repel2;              // repel_from_2
  out[3] = cb_loss;             // cb_loss
  out[4] = relu_mean + cb_loss; // latent_repel_loss
}

// ---------------------------------------------------------------------------
extern "C" void kernel_launch(void* const* d_in, const int* in_sizes, int n_in,
                              void* d_out, int out_size, void* d_ws, size_t ws_size,
                              hipStream_t stream) {
  const float* z  = (const float*)d_in[0];
  const float* cb = (const float*)d_in[1];
  (void)in_sizes; (void)n_in; (void)out_size; (void)ws_size;

  float*    wsf = (float*)d_ws;
  unsigned* wsu = (unsigned*)d_ws;
  float*    out = (float*)d_out;

  float* norms_z = wsf + 2048;
  float* norms_c = wsf + 6144;

  __bf16* Hz = (__bf16*)((char*)d_ws + 40960);
  __bf16* Lz = Hz + NELEM;
  __bf16* Hc = Lz + NELEM;
  __bf16* Lc = Hc + NELEM;

  zero_ws_kernel<<<4, 256, 0, stream>>>(wsu);
  split_bf16_kernel<<<NELEM / 256, 256, 0, stream>>>(z,  Hz, Lz);
  split_bf16_kernel<<<NELEM / 256, 256, 0, stream>>>(cb, Hc, Lc);
  row_norms_kernel<<<N_PTS / 8, 256, 0, stream>>>(z,  norms_z);
  row_norms_kernel<<<N_PTS / 8, 256, 0, stream>>>(cb, norms_c);

  // 256x256 tile grid, 8 waves (tiles) per 256-thread block
  pair_tile_kernel<<<8192, 256, 0, stream>>>(Hz, Lz, norms_z, wsf, wsu, 0);
  quantile_kernel<<<1, 32, 0, stream>>>(wsf, wsu);
  pair_tile_kernel<<<8192, 256, 0, stream>>>(Hz, Lz, norms_z, wsf, wsu, 2);
  pair_tile_kernel<<<8192, 256, 0, stream>>>(Hc, Lc, norms_c, wsf, wsu, 3);

  mid_tile_kernel<<<dim3(512, 10), 256, 0, stream>>>(Hz, Lz, norms_z, wsf);
  finalize_kernel<<<1, 32, 0, stream>>>(wsf, out);
}
